// RNN_transition_74869869904055
// MI455X (gfx1250) — matrix operand order repaired and checked by reading, not compiled, not run on hardware
//
#include <hip/hip_runtime.h>

// Problem constants (from reference): B=64, T=2048, I=H=256
#define M_TOTAL (64 * 2048)        // 131072 independent GRU cells (rows)
#define KDIM    256                // inner dim (I == H == 256)
#define HDIM    256                // output hidden dim
#define WELEMS  (3 * HDIM * KDIM)  // 196608 elements per weight matrix

typedef __attribute__((ext_vector_type(16))) __bf16        v16bf;
typedef __attribute__((ext_vector_type(8)))  float         v8f;
typedef __attribute__((ext_vector_type(4)))  unsigned int  u32x4;
typedef __attribute__((ext_vector_type(2)))  unsigned int  u32x2;

union Frag16 { v16bf v; u32x4 q[2]; };

// Pack two f32 into packed bf16x2 (low = x, high = y).
__device__ __forceinline__ unsigned int pack_bf16x2(float x, float y) {
#if __has_builtin(__builtin_amdgcn_cvt_pk_bf16_f32)
    typedef __attribute__((ext_vector_type(2))) __bf16 v2bf;
    v2bf p = __builtin_amdgcn_cvt_pk_bf16_f32(x, y);
    return __builtin_bit_cast(unsigned int, p);
#else
    // round-half-up to bf16, then splice the two high halves with one v_perm_b32
    unsigned int ux = __float_as_uint(x) + 0x8000u;
    unsigned int uy = __float_as_uint(y) + 0x8000u;
    return __builtin_amdgcn_perm(uy, ux, 0x07060302u);  // [uy[31:16] | ux[31:16]]
#endif
}

__device__ __forceinline__ float sigm(float x) { return 1.0f / (1.0f + __expf(-x)); }
__device__ __forceinline__ float tanh_fast(float x) {
    float a = fabsf(x);
    float e = __expf(-2.0f * a);
    float t = (1.0f - e) / (1.0f + e);
    return copysignf(t, x);
}

// One-shot: convert both weight matrices f32 -> bf16 into workspace (pairwise).
__global__ void cvt_weights_kernel(const float* __restrict__ Wih,
                                   const float* __restrict__ Whh,
                                   unsigned int* __restrict__ wbf) {
    int i = blockIdx.x * blockDim.x + threadIdx.x;
    if (i < WELEMS / 2) {
        float2 a = ((const float2*)Wih)[i];
        float2 b = ((const float2*)Whh)[i];
        wbf[i]              = pack_bf16x2(a.x, a.y);
        wbf[WELEMS / 2 + i] = pack_bf16x2(b.x, b.y);
    }
}

// Fused dual-GEMM + GRU gating.
// Block: 512 threads (16 waves = 4 waves/SIMD -> full 256-VGPR budget, no spills).
// Tile: 64 rows x 128 h-columns.
//   wave w: columns 16*(w&7) (+ column-half offset), row half 32*(w>>3), two 16-row subtiles.
// Row-half partner waves (w, w+8) read identical B fragments -> WGP$ hits.
// blockIdx LSB = column half, so the two WGs sharing an A-tile run concurrently (L2 reuse).
__launch_bounds__(512, 1)
__global__ void gru_cell_wmma_kernel(const float* __restrict__ inp,
                                     const float* __restrict__ h0,
                                     const unsigned short* __restrict__ wih,  // bf16 bits [768,256]
                                     const unsigned short* __restrict__ whh,  // bf16 bits [768,256]
                                     const float* __restrict__ b_ih,
                                     const float* __restrict__ b_hh,
                                     float* __restrict__ out) {
    // Double-buffered bf16 staging of the 64x32 activation tiles.
    // Row pitch 40 bf16 = 80 bytes (16B-aligned, spreads LDS banks). 20 KB total.
    __shared__ __align__(16) unsigned short ldsI[2][64 * 40];
    __shared__ __align__(16) unsigned short ldsH[2][64 * 40];

    const int tid     = threadIdx.x;
    const int wave    = tid >> 5;
    const int lane    = tid & 31;
    const int lr      = lane & 15;
    const int hi      = lane >> 4;           // half-wave select (WMMA lane layout)
    const int rowgrp  = wave >> 3;           // 0 or 1: which 32-row half
    const int colhalf = blockIdx.x & 1;      // which 128-column half of H
    const int m0      = (blockIdx.x >> 1) * 64;              // row tile base
    const int n0      = colhalf * 128 + (wave & 7) * 16;     // h-column base for this wave
    const int h       = n0 + lr;             // this lane's output column

    // ---- LDS fill mapping: one float4 per thread per matrix, fully coalesced ----
    const int fr = tid >> 3;                 // 0..63 row within tile
    const int fc = tid & 7;                  // K float4 index within 32-wide K slab
    const float4* gI = (const float4*)(inp + (size_t)(m0 + fr) * KDIM) + fc;
    const float4* gH = (const float4*)(h0  + (size_t)(m0 + fr) * KDIM) + fc;
    const int ldsOff = fr * 40 + 4 * fc;     // ushort units (8B aligned store)

    auto fill = [&](int buf, int kk) {
        float4 vi = gI[8 * kk];
        float4 vh = gH[8 * kk];
        u32x2 pi = { pack_bf16x2(vi.x, vi.y), pack_bf16x2(vi.z, vi.w) };
        u32x2 ph = { pack_bf16x2(vh.x, vh.y), pack_bf16x2(vh.z, vh.w) };
        *(u32x2*)&ldsI[buf][ldsOff] = pi;
        *(u32x2*)&ldsH[buf][ldsOff] = ph;
    };

    // ---- B-fragment base pointers (bf16 weights, gate g at row h+256g) ----
    // B 32x16 layout: lanes 0-15 hold K=0..7 / 16..23, lanes 16-31 hold K=8..15 / 24..31.
    const unsigned short* pIh[3];
    const unsigned short* pHh[3];
    #pragma unroll
    for (int g = 0; g < 3; ++g) {
        size_t base = (size_t)(h + 256 * g) * KDIM + (size_t)hi * 8;
        pIh[g] = wih + base;
        pHh[g] = whh + base;
    }

    v8f accI[2][3] = {};   // inp-GEMM accumulators: [subtile][gate]
    v8f accH[2][3] = {};   // h0-GEMM accumulators

    int buf = 0;
    fill(0, 0);
    __syncthreads();

    #pragma unroll
    for (int kk = 0; kk < KDIM / 32; ++kk) {
        if (kk + 1 < KDIM / 32) fill(buf ^ 1, kk + 1);

        // A fragments (16-bit A 16x32 layout): lanes 0-15 take K 0..7 / 16..23,
        // lanes 16-31 take K 8..15 / 24..31 -> two ds_load_b128 each.
        Frag16 ai[2], ah[2];
        #pragma unroll
        for (int s = 0; s < 2; ++s) {
            const int row = rowgrp * 32 + 16 * s + lr;
            const unsigned short* pi_ = &ldsI[buf][row * 40];
            const unsigned short* ph_ = &ldsH[buf][row * 40];
            ai[s].q[0] = *(const u32x4*)(pi_ + hi * 8);
            ai[s].q[1] = *(const u32x4*)(pi_ + 16 + hi * 8);
            ah[s].q[0] = *(const u32x4*)(ph_ + hi * 8);
            ah[s].q[1] = *(const u32x4*)(ph_ + 16 + hi * 8);
        }

        #pragma unroll
        for (int g = 0; g < 3; ++g) {
            Frag16 bi, bh;
            const unsigned short* pw0 = pIh[g] + kk * 32;
            bi.q[0] = *(const u32x4*)(pw0);
            bi.q[1] = *(const u32x4*)(pw0 + 16);
            const unsigned short* pw1 = pHh[g] + kk * 32;
            bh.q[0] = *(const u32x4*)(pw1);
            bh.q[1] = *(const u32x4*)(pw1 + 16);

            #pragma unroll
            for (int s = 0; s < 2; ++s) {
                accI[s][g] = __builtin_amdgcn_wmma_f32_16x16x32_bf16(
                    false, ai[s].v, false, bi.v, (short)0, accI[s][g], false, false);
                accH[s][g] = __builtin_amdgcn_wmma_f32_16x16x32_bf16(
                    false, ah[s].v, false, bh.v, (short)0, accH[s][g], false, false);
            }
        }
        __syncthreads();
        buf ^= 1;
    }

    // ---- Fused GRU gating epilogue (f32) ----
    const float bih0 = b_ih[h], bih1 = b_ih[h + 256], bih2 = b_ih[h + 512];
    const float bhh0 = b_hh[h], bhh1 = b_hh[h + 256], bhh2 = b_hh[h + 512];

    #pragma unroll
    for (int s = 0; s < 2; ++s) {
        #pragma unroll
        for (int r = 0; r < 8; ++r) {
            // C/D layout: VGPR r -> row r (lanes 0-15) or row r+8 (lanes 16-31)
            const int m = m0 + rowgrp * 32 + 16 * s + hi * 8 + r;
            float gir = accI[s][0][r] + bih0;
            float giz = accI[s][1][r] + bih1;
            float gin = accI[s][2][r] + bih2;
            float ghr = accH[s][0][r] + bhh0;
            float ghz = accH[s][1][r] + bhh1;
            float ghn = accH[s][2][r] + bhh2;
            float rg = sigm(gir + ghr);
            float zg = sigm(giz + ghz);
            float ng = tanh_fast(gin + rg * ghn);
            float hp = h0[(size_t)m * HDIM + h];      // L2-resident (just streamed)
            out[(size_t)m * HDIM + h] = (1.0f - zg) * ng + zg * hp;
        }
    }
}

extern "C" void kernel_launch(void* const* d_in, const int* in_sizes, int n_in,
                              void* d_out, int out_size, void* d_ws, size_t ws_size,
                              hipStream_t stream) {
    const float* inp = (const float*)d_in[0];   // [B,T,I]
    const float* h0  = (const float*)d_in[1];   // [B,T,H]
    const float* Wih = (const float*)d_in[2];   // [3H,I]
    const float* Whh = (const float*)d_in[3];   // [3H,H]
    const float* bih = (const float*)d_in[4];   // [3H]
    const float* bhh = (const float*)d_in[5];   // [3H]
    float* out = (float*)d_out;

    unsigned int* wbf = (unsigned int*)d_ws;    // bf16 weights: [Wih | Whh] = 768 KB

    cvt_weights_kernel<<<(WELEMS / 2 + 255) / 256, 256, 0, stream>>>(Wih, Whh, wbf);
    // grid = (M tiles of 64 rows) x (2 column halves); column half in LSB
    gru_cell_wmma_kernel<<<(M_TOTAL / 64) * 2, 512, 0, stream>>>(
        inp, h0,
        (const unsigned short*)wbf,
        (const unsigned short*)wbf + WELEMS,
        bih, bhh, out);
}
